// ScaledDotProductAttention_50809463112106
// MI455X (gfx1250) — matrix operand order repaired and checked by reading, not compiled, not run on hardware
//
#include <hip/hip_runtime.h>
#include <math.h>
#include <stdint.h>

typedef __attribute__((ext_vector_type(16))) __bf16 v16bf;
typedef __attribute__((ext_vector_type(8)))  __bf16 v8bf;
typedef __attribute__((ext_vector_type(8)))  float  v8f;
typedef __attribute__((ext_vector_type(4)))  float  v4f;

namespace {
constexpr int BB = 32;        // batch
constexpr int LL = 2048;      // seq len (q and k)
constexpr int DD = 128;       // head dim
constexpr int BQ = 128;       // query rows per workgroup
constexpr int WQ = 16;        // query rows per wave
constexpr int KBL = 32;       // keys per block
constexpr int NW = BQ / WQ;   // 8 waves per workgroup
constexpr int KT_PITCH = 40;  // padded bf16 pitch for K^T tile rows (80B, 16B aligned)
constexpr float INV_TEMP = 1.0f / 11.313708498984761f;
constexpr float NEGINF   = -100000000.0f;
}

__device__ __forceinline__ v16bf join8(v8bf lo, v8bf hi) {
  return __builtin_shufflevector(lo, hi, 0,1,2,3,4,5,6,7,8,9,10,11,12,13,14,15);
}

__global__ __launch_bounds__(256)
void fa_fwd_kernel(const float* __restrict__ Q, const float* __restrict__ K,
                   const float* __restrict__ V, const unsigned char* __restrict__ KPM,
                   float* __restrict__ Out, float* __restrict__ Attn)
{
  // bf16 compute tiles
  __shared__ __align__(16) __bf16 sKt[DD * KT_PITCH];   // K^T tile: [kd][key], padded pitch
  __shared__ __align__(16) __bf16 sV[KBL * DD];         // V tile:   [key][dv]
  __shared__ __align__(16) __bf16 sP[NW * WQ * KBL];    // per-wave P staging (C->A relayout)
  // double-buffered fp32 staging filled by async Global->LDS copies
  __shared__ __align__(16) float sKf[2][KBL * DD];      // 2 x 16 KB
  __shared__ __align__(16) float sVf[2][KBL * DD];      // 2 x 16 KB

  const int b    = blockIdx.y;
  const int qb   = blockIdx.x;
  const int tid  = threadIdx.x;
  const int lane = tid & 31;
  const int wave = tid >> 5;
  const int m    = lane & 15;               // C/D column lane, A row
  const int hb   = (lane >> 4) << 3;        // 0 or 8 (row offset of C/D half)
  const int kl   = m + ((lane >> 4) << 4);  // B-fragment row 0..31

  const int q0   = qb * BQ + wave * WQ;     // this wave's first query row
  const int kend = (qb + 1) * BQ;           // causal: keys iterated in [0, kend)

  const float* Qb = Q + (size_t)b * LL * DD;
  const float* Kb = K + (size_t)b * LL * DD;
  const float* Vb = V + (size_t)b * LL * DD;
  float* attnB = Attn + (size_t)b * LL * LL;
  float* outB  = Out  + (size_t)b * LL * DD;

  // ---- Q as hi/lo bf16 A-fragments (fp32-grade Q via split) ----
  // 16-bit A layout: lanes 0-15 = row m, K = {0..7,16..23}; lanes 16-31 = row m, K = {8..15,24..31}
  v16bf qhi[4], qlo[4];
  {
    const float* qrow = Qb + (size_t)(q0 + m) * DD;
#pragma unroll
    for (int c = 0; c < 4; ++c) {
      const float* p0 = qrow + c * 32 + hb;
      const float* p1 = p0 + 16;
#pragma unroll
      for (int i = 0; i < 8; ++i) {
        float f = p0[i];
        __bf16 h = (__bf16)f;
        qhi[c][i] = h;
        qlo[c][i] = (__bf16)(f - (float)h);
      }
#pragma unroll
      for (int i = 0; i < 8; ++i) {
        float f = p1[i];
        __bf16 h = (__bf16)f;
        qhi[c][i + 8] = h;
        qlo[c][i + 8] = (__bf16)(f - (float)h);
      }
    }
  }

  // key-padding flags for the 8 rows this lane's half covers
  unsigned char kp8[8];
#pragma unroll
  for (int i = 0; i < 8; ++i) kp8[i] = KPM[(size_t)b * LL + q0 + hb + i];

  // ---- async copy of one contiguous 16 KB tile (32 rows x 512 B) into LDS ----
  // 256 threads x 4 x b128 = 16 KB, tracked on ASYNCcnt
  auto issueAsyncTile = [&](const float* gsrc, float* lbuf) {
    uint32_t lbase = (uint32_t)(uintptr_t)lbuf + (uint32_t)tid * 16u;
    unsigned long long gbase = (unsigned long long)(uintptr_t)gsrc + (unsigned long long)tid * 16ull;
#pragma unroll
    for (int r = 0; r < 4; ++r) {
      uint32_t la = lbase + (uint32_t)(r * 4096);
      unsigned long long ga = gbase + (unsigned long long)(r * 4096);
      asm volatile("global_load_async_to_lds_b128 %0, %1, off"
                   :: "v"(la), "v"(ga)
                   : "memory");
    }
  };

  // ---- fp32 (LDS) -> bf16 (LDS) conversion/layout ----
  auto convertKt = [&](const float* lsrc) {      // transpose into K^T tile
    const int key = tid >> 3;                    // 0..31
    const int ch  = (tid & 7) << 4;              // kd chunk: 0,16,..,112
    const float* kp = lsrc + key * DD + ch;
#pragma unroll
    for (int j = 0; j < 16; ++j) sKt[(ch + j) * KT_PITCH + key] = (__bf16)kp[j];
  };
  auto convertVs = [&](const float* lsrc) {      // straight copy
    const int key = tid >> 3;
    const int ch  = (tid & 7) << 4;
    const float* vp = lsrc + key * DD + ch;
#pragma unroll
    for (int j = 0; j < 16; ++j) sV[key * DD + ch + j] = (__bf16)vp[j];
  };

  // S tile (16q x 16k) for key sub-tile n0 in {0,16}: Q(16x128)*K^T(128x16), bf16 split-Q
  auto computeS = [&](int kb, int n0) -> v8f {
    v8f s = {0.f,0.f,0.f,0.f,0.f,0.f,0.f,0.f};
#pragma unroll
    for (int c = 0; c < 4; ++c) {
      const __bf16* bp = &sKt[(c * 32 + hb) * KT_PITCH + n0 + m];
      v16bf bb;
#pragma unroll
      for (int j = 0; j < 8; ++j) {
        bb[j]     = bp[j * KT_PITCH];
        bb[j + 8] = bp[(16 + j) * KT_PITCH];
      }
      s = __builtin_amdgcn_wmma_f32_16x16x32_bf16(false, qhi[c], false, bb, (short)0, s, false, false);
      s = __builtin_amdgcn_wmma_f32_16x16x32_bf16(false, qlo[c], false, bb, (short)0, s, false, false);
    }
#pragma unroll
    for (int i = 0; i < 8; ++i) {
      const int row = q0 + hb + i;
      const int col = kb + n0 + m;
      const bool msk = (col > row) || kp8[i];
      s[i] = msk ? NEGINF : s[i] * INV_TEMP;
    }
    return s;
  };

  // ---- Pass A: online softmax stats (max & sumexp per query row) ----
  float mrun[8], lrun[8];
#pragma unroll
  for (int i = 0; i < 8; ++i) { mrun[i] = -INFINITY; lrun[i] = 0.0f; }

  issueAsyncTile(Kb, sKf[0]);
  int ib = 0;
  for (int kb = 0; kb < kend; kb += KBL) {
    asm volatile("s_wait_asynccnt 0" ::: "memory");
    __syncthreads();                       // tile landed everywhere; prior sKt readers done
    convertKt(sKf[ib]);
    if (kb + KBL < kend)                   // overlap next tile's copy with this tile's math
      issueAsyncTile(Kb + (size_t)(kb + KBL) * DD, sKf[ib ^ 1]);
    __syncthreads();                       // sKt ready
    v8f s0 = computeS(kb, 0);
    v8f s1 = computeS(kb, 16);
#pragma unroll
    for (int i = 0; i < 8; ++i) {
      float mx = fmaxf(s0[i], s1[i]);
#pragma unroll
      for (int off = 1; off < 16; off <<= 1) mx = fmaxf(mx, __shfl_xor(mx, off, 32));
      const float mnew = fmaxf(mrun[i], mx);
      float ps = __expf(s0[i] - mnew) + __expf(s1[i] - mnew);
#pragma unroll
      for (int off = 1; off < 16; off <<= 1) ps += __shfl_xor(ps, off, 32);
      lrun[i] = lrun[i] * __expf(mrun[i] - mnew) + ps;
      mrun[i] = mnew;
    }
    ib ^= 1;
  }

  float linv[8];
#pragma unroll
  for (int i = 0; i < 8; ++i) linv[i] = 1.0f / lrun[i];

  // ---- Pass B: normalized P -> attn (NT stream), P*V -> O ----
  v8f o[8];
#pragma unroll
  for (int t = 0; t < 8; ++t) o[t] = (v8f){0.f,0.f,0.f,0.f,0.f,0.f,0.f,0.f};

  __bf16* pw = &sP[wave * WQ * KBL];

  issueAsyncTile(Kb, sKf[0]);
  issueAsyncTile(Vb, sVf[0]);
  ib = 0;
  for (int kb = 0; kb < kend; kb += KBL) {
    asm volatile("s_wait_asynccnt 0" ::: "memory");
    __syncthreads();
    convertKt(sKf[ib]);
    convertVs(sVf[ib]);
    if (kb + KBL < kend) {
      issueAsyncTile(Kb + (size_t)(kb + KBL) * DD, sKf[ib ^ 1]);
      issueAsyncTile(Vb + (size_t)(kb + KBL) * DD, sVf[ib ^ 1]);
    }
    __syncthreads();
    v8f s0 = computeS(kb, 0);
    v8f s1 = computeS(kb, 16);
#pragma unroll
    for (int i = 0; i < 8; ++i) {
      const float p0 = __expf(s0[i] - mrun[i]) * linv[i];
      const float p1 = __expf(s1[i] - mrun[i]) * linv[i];
      float* arow = attnB + (size_t)(q0 + hb + i) * LL + kb;
      __builtin_nontemporal_store(p0, arow + m);          // 536 MB write-once stream:
      __builtin_nontemporal_store(p1, arow + 16 + m);     // keep it out of L2 (K/V live there)
      pw[(hb + i) * KBL + m]      = (__bf16)p0;           // stage for C->A relayout
      pw[(hb + i) * KBL + 16 + m] = (__bf16)p1;
    }
    asm volatile("s_wait_dscnt 0" ::: "memory");          // wave-internal LDS RAW
    const __bf16* pr = pw + m * KBL;
    v16bf pfrag = join8(*(const v8bf*)(pr + hb), *(const v8bf*)(pr + 16 + hb));
#pragma unroll
    for (int t = 0; t < 8; ++t) {
      const __bf16* vp = &sV[hb * DD + t * 16 + m];
      v16bf vb;
#pragma unroll
      for (int j = 0; j < 8; ++j) {
        vb[j]     = vp[j * DD];
        vb[j + 8] = vp[(16 + j) * DD];
      }
      o[t] = __builtin_amdgcn_wmma_f32_16x16x32_bf16(false, pfrag, false, vb, (short)0, o[t], false, false);
    }
    ib ^= 1;
  }

  // ---- write O (already normalized) ----
#pragma unroll
  for (int t = 0; t < 8; ++t) {
#pragma unroll
    for (int i = 0; i < 8; ++i) {
      __builtin_nontemporal_store(o[t][i], outB + (size_t)(q0 + hb + i) * DD + t * 16 + m);
    }
  }

  // ---- zero-fill attn columns beyond causal range (exact zeros, matches fp32 underflow) ----
  for (int r = 0; r < WQ; ++r) {
    float* rowp = attnB + (size_t)(q0 + r) * LL;
    for (int c = kend + lane * 4; c < LL; c += 128) {
      v4f z = {0.f, 0.f, 0.f, 0.f};
      __builtin_nontemporal_store(z, (v4f*)(rowp + c));
    }
  }
}

extern "C" void kernel_launch(void* const* d_in, const int* in_sizes, int n_in,
                              void* d_out, int out_size, void* d_ws, size_t ws_size,
                              hipStream_t stream) {
  const float* q = (const float*)d_in[0];
  const float* k = (const float*)d_in[1];
  const float* v = (const float*)d_in[2];
  // d_in[3] = attn_mask (triu(k=1)) is applied analytically (causal).
  const unsigned char* kpm = (const unsigned char*)d_in[4];
  float* out  = (float*)d_out;
  float* attn = out + (size_t)BB * LL * DD;   // tuple order: (output, attn)
  dim3 grid(LL / BQ, BB);
  fa_fwd_kernel<<<grid, dim3(256), 0, stream>>>(q, k, v, kpm, out, attn);
}
